// CausalSelfAttention_3607772529012
// MI455X (gfx1250) — matrix operand order repaired and checked by reading, not compiled, not run on hardware
//
#include <hip/hip_runtime.h>
#include <hip/hip_bf16.h>

typedef __attribute__((ext_vector_type(16))) __bf16 v16bf;
typedef __attribute__((ext_vector_type(8)))  __bf16 v8bf;
typedef __attribute__((ext_vector_type(8)))  float  v8f;

#define B_  4
#define T_  2048
#define C_  1024
#define H_  16
#define D_  64

// ---------------------------------------------------------------------------
// CDNA5 async global->LDS copy (ASYNCcnt path, bypasses VGPRs)
// ---------------------------------------------------------------------------
__device__ __forceinline__ void async_copy_b128(void* lds_dst, const void* gsrc) {
  // Generic pointers to LDS carry the LDS byte offset in the low 32 bits.
  uint32_t lds_off = (uint32_t)(uintptr_t)lds_dst;
  asm volatile("global_load_async_to_lds_b128 %0, %1, off"
               :: "v"(lds_off), "v"(gsrc) : "memory");
}
__device__ __forceinline__ void wait_async0() {
  asm volatile("s_wait_asynccnt 0" ::: "memory");
}

// ---------------------------------------------------------------------------
// WMMA helpers (wave32, 16x16x32 bf16 -> f32)
// ---------------------------------------------------------------------------
__device__ __forceinline__ v8f wmma_bf16(v16bf a, v16bf b, v8f c) {
  return __builtin_amdgcn_wmma_f32_16x16x32_bf16(
      false, a, false, b, (short)0, c, false, false);
}

// A fragment: tile row-major [rows][ld] bf16 in LDS.
__device__ __forceinline__ v16bf frag_a(const __bf16* tile, int ld, int r0, int k0) {
  const int lane = threadIdx.x & 31;
  const int half = lane >> 4;
  const __bf16* p = tile + (size_t)(r0 + (lane & 15)) * ld + k0 + half * 8;
  v8bf lo = *(const v8bf*)p;
  v8bf hi = *(const v8bf*)(p + 16);
  return __builtin_shufflevector(lo, hi, 0,1,2,3,4,5,6,7,8,9,10,11,12,13,14,15);
}

// B fragment: tileT is [N][ld] (transposed w.r.t. the GEMM's B).
__device__ __forceinline__ v16bf frag_b(const __bf16* tileT, int ld, int n0, int k0) {
  const int lane = threadIdx.x & 31;
  const int half = lane >> 4;
  const __bf16* p = tileT + (size_t)(n0 + (lane & 15)) * ld + k0 + half * 8;
  v8bf lo = *(const v8bf*)p;
  v8bf hi = *(const v8bf*)(p + 16);
  return __builtin_shufflevector(lo, hi, 0,1,2,3,4,5,6,7,8,9,10,11,12,13,14,15);
}

// ---------------------------------------------------------------------------
// f32 -> bf16 elementwise converter (for x)
// ---------------------------------------------------------------------------
__global__ void cvt_f32_bf16(const float* __restrict__ in, __bf16* __restrict__ out, int n) {
  int i = blockIdx.x * blockDim.x + threadIdx.x;
  int stride = gridDim.x * blockDim.x;
  for (; i < n; i += stride) out[i] = (__bf16)in[i];
}

// ---------------------------------------------------------------------------
// Transpose + convert: in[K][N] f32 -> out[N][K] bf16 (LDS-tiled, coalesced)
// ---------------------------------------------------------------------------
__global__ __launch_bounds__(256) void transpose_cvt(
    const float* __restrict__ in, __bf16* __restrict__ out, int K, int N) {
  __shared__ float tile[32][33];
  const int bx = blockIdx.x * 32;  // N
  const int by = blockIdx.y * 32;  // K
  const int tx = threadIdx.x & 31, ty = threadIdx.x >> 5;  // ty: 0..7
  #pragma unroll
  for (int i = ty; i < 32; i += 8)
    tile[i][tx] = in[(size_t)(by + i) * N + bx + tx];
  __syncthreads();
  #pragma unroll
  for (int i = ty; i < 32; i += 8)
    out[(size_t)(bx + i) * K + by + tx] = (__bf16)tile[tx][i];
}

// ---------------------------------------------------------------------------
// bf16 GEMM: C[M,N] = A[M,K] @ B[K,N] with B pre-transposed as BT[N][K].
// 128x128 block tile, BK=64, async double-buffered LDS, one barrier/K-step.
// 8 waves in a 4x2 grid; each wave = 32x64 output (2x4 WMMA tiles).
// ---------------------------------------------------------------------------
template <bool OUT_F32>
__global__ __launch_bounds__(256) void gemm_bf16_wmma(
    const __bf16* __restrict__ A, const __bf16* __restrict__ BT,
    float* __restrict__ Cf, __bf16* __restrict__ Cb, int M, int N, int K)
{
  __shared__ __bf16 As[2][128][64];   // [buf][M][K]
  __shared__ __bf16 Bs[2][128][64];   // [buf][N][K]

  const int t = threadIdx.x;
  const int m0 = blockIdx.y * 128;
  const int n0 = blockIdx.x * 128;
  const int lane = t & 31, half = lane >> 4, cn = lane & 15;
  const int w = t >> 5;
  const int mband = (w >> 1) * 32;
  const int nband = (w & 1) * 64;

  auto stage = [&](int kstep, int buf) {
    const int k0 = kstep * 64;
    #pragma unroll
    for (int i = 0; i < 4; ++i) {
      int off = (i * 256 + t) * 8;          // 0..8191
      int row = off >> 6, col = off & 63;
      async_copy_b128(&As[buf][row][col], &A [(size_t)(m0 + row) * K + k0 + col]);
      async_copy_b128(&Bs[buf][row][col], &BT[(size_t)(n0 + row) * K + k0 + col]);
    }
  };

  v8f acc[2][4];
  #pragma unroll
  for (int i = 0; i < 2; ++i)
    #pragma unroll
    for (int j = 0; j < 4; ++j)
      #pragma unroll
      for (int r = 0; r < 8; ++r) acc[i][j][r] = 0.0f;

  const int nk = K >> 6;
  stage(0, 0);
  for (int kk = 0; kk < nk; ++kk) {
    wait_async0();
    __syncthreads();
    if (kk + 1 < nk) stage(kk + 1, (kk + 1) & 1);
    const int buf = kk & 1;
    #pragma unroll
    for (int ks = 0; ks < 64; ks += 32) {
      v16bf a0 = frag_a(&As[buf][0][0], 64, mband,      ks);
      v16bf a1 = frag_a(&As[buf][0][0], 64, mband + 16, ks);
      #pragma unroll
      for (int j = 0; j < 4; ++j) {
        v16bf bf = frag_b(&Bs[buf][0][0], 64, nband + 16 * j, ks);
        acc[0][j] = wmma_bf16(a0, bf, acc[0][j]);
        acc[1][j] = wmma_bf16(a1, bf, acc[1][j]);
      }
    }
  }

  #pragma unroll
  for (int i = 0; i < 2; ++i)
    #pragma unroll
    for (int j = 0; j < 4; ++j)
      #pragma unroll
      for (int r = 0; r < 8; ++r) {
        int row = m0 + mband + 16 * i + half * 8 + r;
        int col = n0 + nband + 16 * j + cn;
        float v = acc[i][j][r];
        if (OUT_F32) Cf[(size_t)row * N + col] = v;
        else         Cb[(size_t)row * N + col] = (__bf16)v;
      }
}

// ---------------------------------------------------------------------------
// Flash attention (causal, online softmax). One block per (b, h, 64-row qtile).
// ALL global traffic is async and software-pipelined: K(j+1)/V(j+1) are issued
// right after iteration j's scores are written, overlapping softmax + P@V.
// V lands row-major in Vrow and is transposed LDS->LDS at the top of the next
// iteration. 4 barriers per kv tile.
// ---------------------------------------------------------------------------
__global__ __launch_bounds__(256) void flash_attn_wmma(
    const __bf16* __restrict__ qkv, __bf16* __restrict__ O)
{
  __shared__ __bf16 Qs[64][64];    // [q][d]
  __shared__ __bf16 Ks[64][64];    // [kv][d]  (B-frag layout for Q@K^T)
  __shared__ __bf16 Vrow[64][64];  // [kv][d]  async landing buffer for V
  __shared__ __bf16 Vts[64][64];   // [d][kv]  (B-frag layout for P@V)
  __shared__ __bf16 Ps[64][64];    // [q][kv]
  __shared__ float  Ss[64][64];    // [q][kv]
  __shared__ float  pmax[4][64], psum[4][64];
  __shared__ float  mrow[64], lrow[64], srow[64];

  const int t  = threadIdx.x;
  const int qt = blockIdx.x;
  const int h  = blockIdx.y;
  const int b  = blockIdx.z;
  const int qbase = qt * 64;
  const int lane = t & 31, half = lane >> 4, cn = lane & 15;
  const int w = t >> 5;
  const int qband = (w >> 1) * 16;
  const int cband = (w & 1) * 32;
  const float scale = 0.125f;  // 1/sqrt(64)

  auto stage_kv = [&](int jblk) {
    const int kb = jblk * 64;
    #pragma unroll
    for (int i = 0; i < 2; ++i) {
      int off = (i * 256 + t) * 8;
      int r = off >> 6, c = off & 63;
      size_t base = ((size_t)(b * T_ + kb + r)) * (3 * C_) + h * D_;
      async_copy_b128(&Ks[r][c],   &qkv[base + C_     + c]);
      async_copy_b128(&Vrow[r][c], &qkv[base + 2 * C_ + c]);
    }
  };

  // Prologue: Q, K(0), V(0) all async
  #pragma unroll
  for (int i = 0; i < 2; ++i) {
    int off = (i * 256 + t) * 8;
    int r = off >> 6, c = off & 63;
    async_copy_b128(&Qs[r][c],
        &qkv[((size_t)(b * T_ + qbase + r)) * (3 * C_) + h * D_ + c]);
  }
  stage_kv(0);
  if (t < 64) { mrow[t] = -__builtin_inff(); lrow[t] = 0.0f; }

  v8f accO[2];
  #pragma unroll
  for (int tt = 0; tt < 2; ++tt)
    #pragma unroll
    for (int r = 0; r < 8; ++r) accO[tt][r] = 0.0f;

  for (int j = 0; j <= qt; ++j) {
    const int kbase = j * 64;
    wait_async0();
    __syncthreads();  // Qs/Ks/Vrow for this iteration arrived in all waves

    // LDS->LDS transpose: Vrow[kv][d] -> Vts[d][kv]
    #pragma unroll
    for (int i = 0; i < 2; ++i) {
      int off = (i * 256 + t) * 8;
      int r = off >> 6, c = off & 63;
      v8bf vv = *(const v8bf*)&Vrow[r][c];
      #pragma unroll
      for (int jj = 0; jj < 8; ++jj) Vts[c + jj][r] = vv[jj];
    }

    // S = Q @ K^T
    v8f accS[2];
    #pragma unroll
    for (int tt = 0; tt < 2; ++tt)
      #pragma unroll
      for (int r = 0; r < 8; ++r) accS[tt][r] = 0.0f;
    #pragma unroll
    for (int d0 = 0; d0 < 64; d0 += 32) {
      v16bf a = frag_a(&Qs[0][0], 64, qband, d0);
      accS[0] = wmma_bf16(a, frag_b(&Ks[0][0], 64, cband,      d0), accS[0]);
      accS[1] = wmma_bf16(a, frag_b(&Ks[0][0], 64, cband + 16, d0), accS[1]);
    }
    #pragma unroll
    for (int tt = 0; tt < 2; ++tt)
      #pragma unroll
      for (int r = 0; r < 8; ++r) {
        int row = qband + half * 8 + r;
        int col = cband + 16 * tt + cn;
        float sv = accS[tt][r] * scale;
        Ss[row][col] = ((kbase + col) <= (qbase + row)) ? sv : -1e30f;
      }
    __syncthreads();  // Ss + Vts visible; Ks/Vrow reads complete everywhere

    // Prefetch next K/V tiles (overlaps softmax + P@V below)
    if (j + 1 <= qt) stage_kv(j + 1);

    // Softmax, 4 threads per row (16 cols each)
    const int row = t & 63, q = t >> 6;
    const float mo = mrow[row];  // old max (last write >= 2 barriers ago)
    {
      float pm = -1e30f;
      #pragma unroll
      for (int c = q * 16; c < q * 16 + 16; ++c) pm = fmaxf(pm, Ss[row][c]);
      pmax[q][row] = pm;
    }
    __syncthreads();
    {
      float mt = fmaxf(fmaxf(pmax[0][row], pmax[1][row]),
                       fmaxf(pmax[2][row], pmax[3][row]));
      float mn = fmaxf(mo, mt);
      float sum = 0.0f;
      #pragma unroll
      for (int c = q * 16; c < q * 16 + 16; ++c) {
        float p = __expf(Ss[row][c] - mn);
        Ps[row][c] = (__bf16)p;
        sum += p;
      }
      psum[q][row] = sum;
      if (q == 0) { srow[row] = __expf(mo - mn); mrow[row] = mn; }
    }
    __syncthreads();

    if (t < 64)
      lrow[t] = lrow[t] * srow[t] +
                (psum[0][t] + psum[1][t] + psum[2][t] + psum[3][t]);

    // O = O * srow + P @ V
    #pragma unroll
    for (int tt = 0; tt < 2; ++tt)
      #pragma unroll
      for (int r = 0; r < 8; ++r)
        accO[tt][r] *= srow[qband + half * 8 + r];
    #pragma unroll
    for (int k0 = 0; k0 < 64; k0 += 32) {
      v16bf a = frag_a(&Ps[0][0], 64, qband, k0);
      accO[0] = wmma_bf16(a, frag_b(&Vts[0][0], 64, cband,      k0), accO[0]);
      accO[1] = wmma_bf16(a, frag_b(&Vts[0][0], 64, cband + 16, k0), accO[1]);
    }
    // no end-of-loop barrier needed: next iteration's top barrier orders
    // PV reads vs. the following transpose / S / softmax writes.
  }

  __syncthreads();  // publish final lrow to all threads

  #pragma unroll
  for (int tt = 0; tt < 2; ++tt)
    #pragma unroll
    for (int r = 0; r < 8; ++r) {
      int row = qband + half * 8 + r;
      int col = cband + 16 * tt + cn;
      float ov = accO[tt][r] / lrow[row];
      O[((size_t)(b * T_ + qbase + row)) * C_ + h * D_ + col] = (__bf16)ov;
    }
}

// ---------------------------------------------------------------------------
// Launch
// ---------------------------------------------------------------------------
extern "C" void kernel_launch(void* const* d_in, const int* in_sizes, int n_in,
                              void* d_out, int out_size, void* d_ws, size_t ws_size,
                              hipStream_t stream) {
  (void)in_sizes; (void)n_in; (void)out_size; (void)ws_size;
  const float* x    = (const float*)d_in[0];
  const float* Wqkv = (const float*)d_in[1];   // [C, 3C]
  const float* Wout = (const float*)d_in[2];   // [C, C]
  float* out = (float*)d_out;

  const size_t nX   = (size_t)B_ * T_ * C_;
  const size_t nWq  = (size_t)C_ * 3 * C_;
  const size_t nWo  = (size_t)C_ * C_;
  const size_t nQKV = (size_t)B_ * T_ * 3 * C_;

  char* p = (char*)d_ws;
  __bf16* xb    = (__bf16*)p; p += nX   * 2;
  __bf16* wqbT  = (__bf16*)p; p += nWq  * 2;   // [3C][C]
  __bf16* wobT  = (__bf16*)p; p += nWo  * 2;   // [C][C] transposed
  __bf16* qkvb  = (__bf16*)p; p += nQKV * 2;
  __bf16* ob    = (__bf16*)p; p += nX   * 2;

  cvt_f32_bf16<<<1024, 256, 0, stream>>>(x, xb, (int)nX);
  transpose_cvt<<<dim3(3 * C_ / 32, C_ / 32), 256, 0, stream>>>(Wqkv, wqbT, C_, 3 * C_);
  transpose_cvt<<<dim3(C_ / 32, C_ / 32),     256, 0, stream>>>(Wout, wobT, C_, C_);

  // qkv = x @ W_qkv (bf16 out)
  gemm_bf16_wmma<false><<<dim3((3 * C_) / 128, (B_ * T_) / 128), 256, 0, stream>>>(
      xb, wqbT, nullptr, qkvb, B_ * T_, 3 * C_, C_);

  // attention
  flash_attn_wmma<<<dim3(T_ / 64, H_, B_), 256, 0, stream>>>(qkvb, ob);

  // out = O @ W_out (f32 out)
  gemm_bf16_wmma<true><<<dim3(C_ / 128, (B_ * T_) / 128), 256, 0, stream>>>(
      ob, wobT, out, nullptr, B_ * T_, C_, C_);
}
